// MSDNContext_78812649881724
// MI455X (gfx1250) — matrix operand order repaired and checked by reading, not compiled, not run on hardware
//
#include <hip/hip_runtime.h>
#include <hip/hip_bf16.h>
#include <utility>

// ---------------------------------------------------------------------------
// MSDN forward for MI455X (gfx1250, wave32, WMMA).
//  - all M x 512 GEMMs: LDS-tiled bf16 WMMA, 128x64 block tile, 8 waves,
//    each wave 16x64 (4 accumulators, A-fragment reuse)
//  - gate path: 16 rows/block, LN fused, 1024x64 GEMM on WMMA against a
//    pre-transposed bf16 copy of gw (fragments = contiguous 32B loads)
// ---------------------------------------------------------------------------

typedef __attribute__((ext_vector_type(16))) __bf16 v16bf;
typedef __attribute__((ext_vector_type(8)))  __bf16 v8bf;
typedef __attribute__((ext_vector_type(8)))  float  v8f;

#define HID    512
#define GWIDTH 64
#define N_OBJ  4096
#define N_REL  65536

#define GF_RELU_A   1
#define GF_ACCUM    2
#define GF_BIAS     4
#define GF_RELU_OUT 8

#define LDS_STRIDE 40   // 32 payload + 8 pad (bf16) -> 80B row, spreads banks

__device__ __forceinline__ void cvt8_store(float4 f0, float4 f1, bool relu,
                                           float alpha, __bf16* dst)
{
    float v[8] = {f0.x, f0.y, f0.z, f0.w, f1.x, f1.y, f1.z, f1.w};
    v8bf o;
#pragma unroll
    for (int i = 0; i < 8; ++i) {
        float x = v[i];
        if (relu) x = fmaxf(x, 0.f) * alpha;
        o[i] = (__bf16)x;
    }
    *(v8bf*)dst = o;
}

__device__ __forceinline__ v16bf cat16(v8bf lo, v8bf hi)
{
    v16bf r;
#pragma unroll
    for (int i = 0; i < 8; ++i) { r[i] = lo[i]; r[8 + i] = hi[i]; }
    return r;
}

// ---------------------------------------------------------------------------
// C[M,N] (+)= (alpha*relu?(A)) @ W[K,N] + bias   (M%128==0, N%64==0, K%32==0)
// grid = (N/64, M/128), block = 256
// ---------------------------------------------------------------------------
__global__ __launch_bounds__(256)
void gemm_bf16_wmma(const float* __restrict__ A, const float* __restrict__ W,
                    const float* __restrict__ bias, float* __restrict__ C,
                    int M, int N, int K, int flags, float alpha)
{
    __shared__ __bf16 lsA[128 * LDS_STRIDE];   // [row][k] 10KB
    __shared__ __bf16 lsB[64 * LDS_STRIDE];    // [col][k] 5KB (transposed)

    const int t    = threadIdx.x;
    const int w    = t >> 5;
    const int lane = t & 31;
    const int half = lane >> 4;
    const int l15  = lane & 15;
    const int rowB = blockIdx.y * 128;
    const int colB = blockIdx.x * 64;
    const bool relu_a = (flags & GF_RELU_A) != 0;

    // staging coordinates (A: 512 groups of 8 floats; B: 256 groups)
    const int ar0 = t >> 2,          ak0 = (t & 3) * 8;
    const int ar1 = (t + 256) >> 2,  ak1 = ((t + 256) & 3) * 8;
    const int bk  = t & 31,          bn8 = t >> 5;

    v8f acc[4] = {};

    for (int k0 = 0; k0 < K; k0 += 32) {
        // ---- global loads into registers (overlaps prior compute)
        const float* pa0 = A + (size_t)(rowB + ar0) * K + k0 + ak0;
        const float* pa1 = A + (size_t)(rowB + ar1) * K + k0 + ak1;
        float4 a00 = *(const float4*)pa0, a01 = *(const float4*)(pa0 + 4);
        float4 a10 = *(const float4*)pa1, a11 = *(const float4*)(pa1 + 4);
        const float* pb = W + (size_t)(k0 + bk) * N + colB + bn8 * 8;
        float4 b0 = *(const float4*)pb, b1 = *(const float4*)(pb + 4);
        __builtin_prefetch(pa0 + 64, 0, 0);      // streaming A (HBM roofline term)

        __syncthreads();                         // prior compute done reading LDS
        cvt8_store(a00, a01, relu_a, alpha, &lsA[ar0 * LDS_STRIDE + ak0]);
        cvt8_store(a10, a11, relu_a, alpha, &lsA[ar1 * LDS_STRIDE + ak1]);
        {
            float bv[8] = {b0.x, b0.y, b0.z, b0.w, b1.x, b1.y, b1.z, b1.w};
#pragma unroll
            for (int i = 0; i < 8; ++i)
                lsB[(bn8 * 8 + i) * LDS_STRIDE + bk] = (__bf16)bv[i];
        }
        __syncthreads();                         // tiles visible

        // ---- fragments: two contiguous 8-elem bf16 chunks each (2x b128)
        const int arow = (w * 16 + l15) * LDS_STRIDE;
        v16bf af = cat16(*(const v8bf*)&lsA[arow + half * 16],
                         *(const v8bf*)&lsA[arow + half * 16 + 8]);
#pragma unroll
        for (int s = 0; s < 4; ++s) {
            const __bf16* bp = &lsB[(s * 16 + l15) * LDS_STRIDE + half * 16];
            v16bf bf_ = cat16(*(const v8bf*)bp, *(const v8bf*)(bp + 8));
            acc[s] = __builtin_amdgcn_wmma_f32_16x16x32_bf16(
                false, af, false, bf_, (short)0, acc[s], false, false);
        }
    }

    // ---- epilogue: D row = v + 8*half, col = l15 (per 16-col subtile)
    const int wrow0 = rowB + w * 16;
#pragma unroll
    for (int s = 0; s < 4; ++s) {
        const int col = colB + s * 16 + l15;
        const float b = (flags & GF_BIAS) ? bias[col] : 0.f;
#pragma unroll
        for (int v = 0; v < 8; ++v) {
            float* cp = C + (size_t)(wrow0 + v + half * 8) * N + col;
            float val = acc[s][v] + b;
            if (flags & GF_ACCUM)    val += *cp;
            if (flags & GF_RELU_OUT) val = fmaxf(val, 0.f);
            *cp = val;
        }
    }
}

// ---------------------------------------------------------------------------
// gw transpose + bf16 convert:  gwT[col][k] = bf16(gw[k][col]),  K=1024, C=64
// ---------------------------------------------------------------------------
__global__ __launch_bounds__(256)
void transpose_gw(const float* __restrict__ gw, __bf16* __restrict__ gwT)
{
    const int i = blockIdx.x * 256 + threadIdx.x;   // 65536
    const int k = i >> 6, c = i & 63;
    gwT[c * 1024 + k] = (__bf16)gw[i];
}

// ---------------------------------------------------------------------------
// Fused gate, 16 relation rows per 256-thread block:
//   x = concat(unary,pair); h = relu(LN(x)); g = mean(sigmoid(h@gw+gb));
//   out (+)= pair * g
// gwT is the bf16 [col][k] transpose (B fragments = contiguous 32B, L2-hot).
// ---------------------------------------------------------------------------
#define XH_STRIDE 1032   // 1024 + 8 pad bf16

__global__ __launch_bounds__(256)
void gate16_wmma(const float* __restrict__ ubase, const int* __restrict__ uidx,
                 const float* __restrict__ pbase, const int* __restrict__ pidx,
                 const float* __restrict__ lng, const float* __restrict__ lnb,
                 const __bf16* __restrict__ gwT, const float* __restrict__ gb,
                 float* __restrict__ out, int accum)
{
    __shared__ __bf16 xh[16 * XH_STRIDE];    // 33KB: relu(LN(x)) in bf16
    __shared__ float  part[16 * GWIDTH];     // 4KB: cross-wave partials
    __shared__ float  red[256];
    __shared__ float  mu[16], rsd[16], gs[16];
    __shared__ int    ur[16], pr[16];

    const int t   = threadIdx.x;
    const int row = t & 15;
    const int sub = t >> 4;
    const int r0  = blockIdx.x * 16;

    if (t < 16) {
        ur[t] = uidx ? uidx[2 * (r0 + t)] : (r0 + t);
        pr[t] = pidx ? pidx[2 * (r0 + t)] : (r0 + t);
    }
    for (int i = t; i < 16 * GWIDTH; i += 256) part[i] = 0.f;
    __syncthreads();

    const float* u = ubase + (size_t)ur[row] * HID;
    const float* p = pbase + (size_t)pr[row] * HID;

    // ---- LN stats (each of 16 subs covers 32 u-elems + 32 p-elems)
    float s = 0.f, ss = 0.f;
    for (int i = 0; i < 32; ++i) { const float v = u[sub * 32 + i]; s += v; ss += v * v; }
    for (int i = 0; i < 32; ++i) { const float v = p[sub * 32 + i]; s += v; ss += v * v; }
    red[t] = s; __syncthreads();
    for (int o = 8; o >= 1; o >>= 1) { if (sub < o) red[t] += red[t + o * 16]; __syncthreads(); }
    if (sub == 0) mu[row] = red[row] * (1.f / 1024.f);
    __syncthreads();
    red[t] = ss; __syncthreads();
    for (int o = 8; o >= 1; o >>= 1) { if (sub < o) red[t] += red[t + o * 16]; __syncthreads(); }
    if (sub == 0) {
        const float m = mu[row];
        rsd[row] = rsqrtf(red[row] * (1.f / 1024.f) - m * m + 1e-5f);
    }
    __syncthreads();
    const float m_ = mu[row], r_ = rsd[row];

    // ---- LN + relu -> bf16 LDS (vectorized 16B stores)
#pragma unroll
    for (int c = 0; c < 4; ++c) {
        const int j0 = sub * 32 + c * 8;
        v8bf o;
#pragma unroll
        for (int i = 0; i < 8; ++i) {
            const float y = (u[j0 + i] - m_) * r_ * lng[j0 + i] + lnb[j0 + i];
            o[i] = (__bf16)fmaxf(y, 0.f);
        }
        *(v8bf*)&xh[row * XH_STRIDE + j0] = o;
    }
#pragma unroll
    for (int c = 0; c < 4; ++c) {
        const int j0 = sub * 32 + c * 8, jj = HID + j0;
        v8bf o;
#pragma unroll
        for (int i = 0; i < 8; ++i) {
            const float y = (p[j0 + i] - m_) * r_ * lng[jj + i] + lnb[jj + i];
            o[i] = (__bf16)fmaxf(y, 0.f);
        }
        *(v8bf*)&xh[row * XH_STRIDE + jj] = o;
    }
    __syncthreads();

    // ---- WMMA: 8 waves split K=1024 into 128-wide chunks
    const int w = t >> 5, lane = t & 31, half = lane >> 4, l15 = lane & 15;
    v8f acc[4] = {};
#pragma unroll
    for (int ks = 0; ks < 4; ++ks) {
        const int kk = w * 128 + ks * 32;
        v16bf af = cat16(*(const v8bf*)&xh[l15 * XH_STRIDE + kk + half * 16],
                         *(const v8bf*)&xh[l15 * XH_STRIDE + kk + half * 16 + 8]);
#pragma unroll
        for (int s4 = 0; s4 < 4; ++s4) {
            const __bf16* bp = gwT + (size_t)(s4 * 16 + l15) * 1024 + kk + half * 16;
            v16bf bf_ = cat16(*(const v8bf*)bp, *(const v8bf*)(bp + 8));
            acc[s4] = __builtin_amdgcn_wmma_f32_16x16x32_bf16(
                false, af, false, bf_, (short)0, acc[s4], false, false);
        }
    }
#pragma unroll
    for (int s4 = 0; s4 < 4; ++s4)
#pragma unroll
        for (int v = 0; v < 8; ++v)
            atomicAdd(&part[(v + half * 8) * GWIDTH + s4 * 16 + l15], acc[s4][v]);
    __syncthreads();

    // ---- bias + sigmoid
    for (int i = t; i < 16 * GWIDTH; i += 256) {
        const float d = part[i] + gb[i & 63];
        part[i] = 1.f / (1.f + __expf(-d));
    }
    __syncthreads();
    // ---- per-row mean of 64 -> gate scalar
    {
        float a = 0.f;
#pragma unroll
        for (int i = 0; i < 4; ++i) a += part[row * GWIDTH + sub * 4 + i];
        red[t] = a; __syncthreads();
        for (int o = 8; o >= 1; o >>= 1) { if (sub < o) red[t] += red[t + o * 16]; __syncthreads(); }
        if (sub == 0) gs[row] = red[row] * (1.f / 64.f);
    }
    __syncthreads();

    const float g = gs[row];
    float* orow = out + (size_t)(r0 + row) * HID;
#pragma unroll
    for (int c = 0; c < 8; ++c) {
        const int j = sub * 32 + c * 4;
        float4 pv = *(const float4*)&p[j];
        float4 ov;
        ov.x = pv.x * g; ov.y = pv.y * g; ov.z = pv.z * g; ov.w = pv.w * g;
        if (accum) {
            float4 cur = *(const float4*)&orow[j];
            ov.x += cur.x; ov.y += cur.y; ov.z += cur.z; ov.w += cur.w;
        }
        *(float4*)&orow[j] = ov;
    }
}

// ---------------------------------------------------------------------------
// Segment sum + count via f32 global atomics; then combine to mean.
// ---------------------------------------------------------------------------
__global__ __launch_bounds__(256)
void seg_accum(const float* __restrict__ vals, const int* __restrict__ idx,
               float* __restrict__ sum, float* __restrict__ cnt)
{
    const int r = blockIdx.x;
    const int t = threadIdx.x;
    const int seg = idx[2 * r];
    const float* v = vals + (size_t)r * HID;
    float* srow = sum + (size_t)seg * HID;
    for (int i = t; i < HID; i += 256) atomicAdd(&srow[i], v[i]);
    if (t == 0) atomicAdd(&cnt[seg], 1.0f);
}

__global__ __launch_bounds__(256)
void seg_combine(const float* __restrict__ ssub, const float* __restrict__ csub,
                 const float* __restrict__ sobj, const float* __restrict__ cobj,
                 float* __restrict__ m)
{
    const long long i = (long long)blockIdx.x * 256 + threadIdx.x;
    if (i >= (long long)N_OBJ * HID) return;
    const int row = (int)(i >> 9);
    const float cs = fmaxf(csub[row], 1.f);
    const float co = fmaxf(cobj[row], 1.f);
    m[i] = 0.5f * (ssub[i] / cs + sobj[i] / co);
}

__global__ __launch_bounds__(256)
void fill_zero(float* __restrict__ p, long long n)
{
    const long long i = (long long)blockIdx.x * 256 + threadIdx.x;
    if (i < n) p[i] = 0.f;
}

// ---------------------------------------------------------------------------
// Host orchestration
// ---------------------------------------------------------------------------
extern "C" void kernel_launch(void* const* d_in, const int* in_sizes, int n_in,
                              void* d_out, int out_size, void* d_ws, size_t ws_size,
                              hipStream_t stream)
{
    (void)in_sizes; (void)n_in; (void)out_size; (void)ws_size;

    const float* box  = (const float*)d_in[0];
    const float* uni  = (const float*)d_in[1];
    const int*   rp   = (const int*)d_in[2];
    const float* wobj = (const float*)d_in[3];
    const float* bobj = (const float*)d_in[4];
    const float* wrel = (const float*)d_in[5];
    const float* brel = (const float*)d_in[6];
    const float* lng[4]; const float* lnb[4]; const float* gw[4]; const float* gb[4];
    for (int g = 0; g < 4; ++g) {              // p2s, p2o, s2p, o2p
        lng[g] = (const float*)d_in[7 + 4 * g + 0];
        lnb[g] = (const float*)d_in[7 + 4 * g + 1];
        gw[g]  = (const float*)d_in[7 + 4 * g + 2];
        gb[g]  = (const float*)d_in[7 + 4 * g + 3];
    }
    const float* wih_o  = (const float*)d_in[23];
    const float* bih_o  = (const float*)d_in[24];
    const float* whh_o  = (const float*)d_in[25];
    const float* bhh_o  = (const float*)d_in[26];
    const float* wih_pr = (const float*)d_in[27];
    const float* bih_pr = (const float*)d_in[28];
    const float* whh_pr = (const float*)d_in[29];
    const float* bhh_pr = (const float*)d_in[30];

    const int* subj = rp;
    const int* obj  = rp + 1;

    float* ws = (float*)d_ws;
    size_t off = 0;
    auto carve = [&](size_t n) { float* q = ws + off; off += n; return q; };
    float* inst_a = carve((size_t)N_OBJ * HID);
    float* inst_b = carve((size_t)N_OBJ * HID);
    float* rel_a  = carve((size_t)N_REL * HID);
    float* rel_b  = carve((size_t)N_REL * HID);
    float* g1     = carve((size_t)N_REL * HID);
    float* ssub   = carve((size_t)N_OBJ * HID);   // ssub,sobj,csub,cobj contiguous
    float* sobj   = carve((size_t)N_OBJ * HID);
    float* csub   = carve((size_t)N_OBJ);
    float* cobj   = carve((size_t)N_OBJ);
    float* mbuf   = carve((size_t)N_OBJ * HID);
    __bf16* gwt[4];
    for (int g = 0; g < 4; ++g) gwt[g] = (__bf16*)carve(1024 * GWIDTH / 2);

    const long long seg_zero_n = 2LL * N_OBJ * HID + 2LL * N_OBJ;
    const size_t inst_bytes = (size_t)N_OBJ * HID * sizeof(float);
    const size_t rel_bytes  = (size_t)N_REL * HID * sizeof(float);

    // gw transposes (bf16, [col][k]) once per launch
    for (int g = 0; g < 4; ++g)
        transpose_gw<<<1024 * GWIDTH / 256, 256, 0, stream>>>(gw[g], gwt[g]);

    // input projections
    gemm_bf16_wmma<<<dim3(HID / 64, N_OBJ / 128), 256, 0, stream>>>(
        box, wobj, bobj, inst_a, N_OBJ, HID, 4096, GF_BIAS | GF_RELU_OUT, 1.f);
    gemm_bf16_wmma<<<dim3(HID / 64, N_REL / 128), 256, 0, stream>>>(
        uni, wrel, brel, rel_a, N_REL, HID, 4096, GF_BIAS | GF_RELU_OUT, 1.f);

    float* ci = inst_a; float* ni = inst_b;
    float* cr = rel_a;  float* nr = rel_b;

    for (int it = 0; it < 2; ++it) {
        fill_zero<<<(int)((seg_zero_n + 255) / 256), 256, 0, stream>>>(ssub, seg_zero_n);

        gate16_wmma<<<N_REL / 16, 256, 0, stream>>>(ci, subj, cr, nullptr,
            lng[0], lnb[0], gwt[0], gb[0], g1, 0);
        seg_accum<<<N_REL, 256, 0, stream>>>(g1, subj, ssub, csub);
        gate16_wmma<<<N_REL / 16, 256, 0, stream>>>(ci, obj, cr, nullptr,
            lng[1], lnb[1], gwt[1], gb[1], g1, 0);
        seg_accum<<<N_REL, 256, 0, stream>>>(g1, obj, sobj, cobj);
        seg_combine<<<(int)(((long long)N_OBJ * HID + 255) / 256), 256, 0, stream>>>(
            ssub, csub, sobj, cobj, mbuf);

        hipMemcpyAsync(ni, ci, inst_bytes, hipMemcpyDeviceToDevice, stream);
        gemm_bf16_wmma<<<dim3(HID / 64, N_OBJ / 128), 256, 0, stream>>>(
            mbuf, wih_o, bih_o, ni, N_OBJ, HID, HID, GF_RELU_A | GF_ACCUM | GF_BIAS, 1.f);
        gemm_bf16_wmma<<<dim3(HID / 64, N_OBJ / 128), 256, 0, stream>>>(
            ci, whh_o, bhh_o, ni, N_OBJ, HID, HID, GF_RELU_A | GF_ACCUM | GF_BIAS, 1.f);

        gate16_wmma<<<N_REL / 16, 256, 0, stream>>>(cr, nullptr, ci, subj,
            lng[2], lnb[2], gwt[2], gb[2], g1, 0);
        gate16_wmma<<<N_REL / 16, 256, 0, stream>>>(cr, nullptr, ci, obj,
            lng[3], lnb[3], gwt[3], gb[3], g1, 1);

        hipMemcpyAsync(nr, cr, rel_bytes, hipMemcpyDeviceToDevice, stream);
        gemm_bf16_wmma<<<dim3(HID / 64, N_REL / 128), 256, 0, stream>>>(
            g1, wih_pr, bih_pr, nr, N_REL, HID, HID, GF_RELU_A | GF_ACCUM | GF_BIAS, 0.5f);
        gemm_bf16_wmma<<<dim3(HID / 64, N_REL / 128), 256, 0, stream>>>(
            cr, whh_pr, bhh_pr, nr, N_REL, HID, HID, GF_RELU_A | GF_ACCUM | GF_BIAS, 1.f);

        std::swap(ci, ni);
        std::swap(cr, nr);
    }

    float* out = (float*)d_out;
    hipMemcpyAsync(out, ci, inst_bytes, hipMemcpyDeviceToDevice, stream);
    hipMemcpyAsync(out + (size_t)N_OBJ * HID, cr, rel_bytes, hipMemcpyDeviceToDevice, stream);
}
